// PointNet2_Classifier_44693429682246
// MI455X (gfx1250) — compile-verified, matrix operand order checked by reading, not run on hardware
//
#include <hip/hip_runtime.h>
#include <hip/hip_bf16.h>

typedef _Float16 half_t;
typedef __attribute__((ext_vector_type(8)))  _Float16 v8h;
typedef __attribute__((ext_vector_type(16))) _Float16 v16h;
typedef __attribute__((ext_vector_type(8)))  float    v8f;

#define DEV_INLINE __device__ __forceinline__
#define CDIV(a,b) (((a)+(b)-1)/(b))

// ---------------- geometry / sizes ----------------
// B=16, N=4096 ; M1=ceil(.1*4096)=410, M2=ceil(.2*410)=82, M3=ceil(.5*82)=41
// stage K: 16,32,64 per radius; edge rows: 104960 / 41984 / 41984 (all %16==0)

// ---------------- workspace layout ----------------
static constexpr size_t alignup(size_t x){ return (x + 255) & ~(size_t)255; }

// f16 transposed/padded weights, offsets in halfs
static constexpr size_t O_W10 = 0;                       // 16 x 32
static constexpr size_t O_W11 = O_W10 + 16*32;           // 32 x 32
static constexpr size_t O_W20 = O_W11 + 32*32;           // 64 x 96
static constexpr size_t O_W21 = O_W20 + 64*96;           // 128 x 64
static constexpr size_t O_W30 = O_W21 + 128*64;          // 256 x 288
static constexpr size_t O_W31 = O_W30 + 256*288;         // 512 x 256
static constexpr size_t O_W40 = O_W31 + 512*256;         // 1024 x 1056
static constexpr size_t O_W41 = O_W40 + (size_t)1024*1056; // 2048 x 1024
static constexpr size_t O_H1  = O_W41 + (size_t)2048*1024; // 512 x 2048
static constexpr size_t O_H2  = O_H1  + (size_t)512*2048;  // 64 x 512
static constexpr size_t WT_HALFS = O_H2 + 64*512;          // 4,480,512

static constexpr size_t OFF_WT   = 0;
static constexpr size_t OFF_IDX1 = alignup(OFF_WT   + WT_HALFS*2);
static constexpr size_t OFF_Q1   = alignup(OFF_IDX1 + (size_t)16*410*4);
static constexpr size_t OFF_IDX2 = alignup(OFF_Q1   + (size_t)16*410*3*4);
static constexpr size_t OFF_Q2   = alignup(OFF_IDX2 + (size_t)16*82*4);
static constexpr size_t OFF_IDX3 = alignup(OFF_Q2   + (size_t)16*82*3*4);
static constexpr size_t OFF_Q3   = alignup(OFF_IDX3 + (size_t)16*41*4);
static constexpr size_t OFF_X1   = alignup(OFF_Q3   + (size_t)16*41*3*4);
static constexpr size_t OFF_X2   = alignup(OFF_X1   + (size_t)16*410*64*4);
static constexpr size_t OFF_X3   = alignup(OFF_X2   + (size_t)16*82*256*4);
static constexpr size_t OFF_NIDX = alignup(OFF_X3   + (size_t)16*41*1024*4);
static constexpr size_t OFF_VALID= alignup(OFF_NIDX + (size_t)104960*4);
static constexpr size_t OFF_STATS= alignup(OFF_VALID+ (size_t)104960);
static constexpr size_t OFF_HA   = alignup(OFF_STATS+ (size_t)4098*4);
static constexpr size_t OFF_HB   = alignup(OFF_HA   + (size_t)21495808*2); // 41984x512 halfs
static constexpr size_t WS_NEED  = alignup(OFF_HB   + (size_t)10747904*2); // 41984x256 halfs

// ---------------- tiny utility kernels ----------------
__global__ void k_zero_f32(float* p, int n){
  int i = blockIdx.x*blockDim.x + threadIdx.x;
  if (i < n) p[i] = 0.f;
}
__global__ void k_set_f32(float* p, float v){
  if (blockIdx.x==0 && threadIdx.x==0) p[0] = v;
}
__global__ void k_zero_f16(half_t* p, int n){
  int i = blockIdx.x*blockDim.x + threadIdx.x;
  if (i < n) p[i] = (half_t)0.f;
}
__global__ void k_count_valid(const unsigned char* v, int n, float* out){
  __shared__ float sh[256];
  int i = blockIdx.x*blockDim.x + threadIdx.x;
  sh[threadIdx.x] = (i < n) ? (float)v[i] : 0.f;
  __syncthreads();
  for (int s = 128; s > 0; s >>= 1){
    if (threadIdx.x < s) sh[threadIdx.x] += sh[threadIdx.x + s];
    __syncthreads();
  }
  if (threadIdx.x == 0) atomicAdd(out, sh[0]);
}
// W [Kin,Nout] f32 -> Wt [Nout,Kpad] f16 (transposed, K zero-padded)
__global__ void k_wt_conv(const float* __restrict__ W, half_t* __restrict__ Wt,
                          int Kin, int Nout, int Kpad){
  int i = blockIdx.x*blockDim.x + threadIdx.x;
  if (i >= Nout*Kpad) return;
  int n = i / Kpad, k = i - n*Kpad;
  Wt[i] = (k < Kin) ? (half_t)W[(size_t)k*Nout + n] : (half_t)0.f;
}

// ---------------- farthest point sampling ----------------
// one block per cloud; min-distance array lives in LDS (N<=4096 -> 16KB)
__global__ void k_fps(const float* __restrict__ pts, int N, int M,
                      int* __restrict__ idx, float* __restrict__ q){
  __shared__ float mind[4096];
  __shared__ float rv[256];
  __shared__ int   ri[256];
  __shared__ float lastp[3];
  int b = blockIdx.x, t = threadIdx.x, bs = blockDim.x;
  const float* P = pts + (size_t)b*N*3;
  for (int i = t; i < N; i += bs) mind[i] = 3.4e38f;
  if (t == 0){
    idx[(size_t)b*M] = 0;
    lastp[0] = P[0]; lastp[1] = P[1]; lastp[2] = P[2];
    q[(size_t)b*M*3+0] = P[0]; q[(size_t)b*M*3+1] = P[1]; q[(size_t)b*M*3+2] = P[2];
  }
  __syncthreads();
  for (int m = 1; m < M; m++){
    float lx = lastp[0], ly = lastp[1], lz = lastp[2];
    float bv = -1.f; int bi = N;
    for (int i = t; i < N; i += bs){
      float dx = P[3*i]-lx, dy = P[3*i+1]-ly, dz = P[3*i+2]-lz;
      float d  = dx*dx + dy*dy + dz*dz;
      float nm = fminf(mind[i], d);
      mind[i] = nm;
      if (nm > bv || (nm == bv && i < bi)){ bv = nm; bi = i; }
    }
    rv[t] = bv; ri[t] = bi;
    __syncthreads();
    for (int s = bs >> 1; s > 0; s >>= 1){
      if (t < s){
        if (rv[t+s] > rv[t] || (rv[t+s] == rv[t] && ri[t+s] < ri[t])){
          rv[t] = rv[t+s]; ri[t] = ri[t+s];
        }
      }
      __syncthreads();
    }
    if (t == 0){
      int w = ri[0];
      idx[(size_t)b*M + m] = w;
      lastp[0] = P[3*w]; lastp[1] = P[3*w+1]; lastp[2] = P[3*w+2];
      q[((size_t)b*M+m)*3+0] = lastp[0];
      q[((size_t)b*M+m)*3+1] = lastp[1];
      q[((size_t)b*M+m)*3+2] = lastp[2];
    }
    __syncthreads();
  }
}

// ---------------- ball query: K nearest within radius ----------------
__global__ void k_ballq(const float* __restrict__ q, const float* __restrict__ p,
                        int M, int N, float r2, int K,
                        int* __restrict__ nidx, unsigned char* __restrict__ valid){
  int gid = blockIdx.x*blockDim.x + threadIdx.x;
  if (gid >= 16*M) return;
  int b = gid / M;
  float qx = q[3*(size_t)gid], qy = q[3*(size_t)gid+1], qz = q[3*(size_t)gid+2];
  const float* P = p + (size_t)b*N*3;
  float kd[64]; int ki[64]; int cnt = 0;
  for (int j = 0; j < N; j++){
    float dx = P[3*j]-qx, dy = P[3*j+1]-qy, dz = P[3*j+2]-qz;
    float d2 = dx*dx + dy*dy + dz*dz;
    if (d2 <= r2){
      if (cnt < K){
        int pos = cnt++;
        while (pos > 0 && kd[pos-1] > d2){ kd[pos]=kd[pos-1]; ki[pos]=ki[pos-1]; pos--; }
        kd[pos] = d2; ki[pos] = j;
      } else if (d2 < kd[K-1]){
        int pos = K-1;
        while (pos > 0 && kd[pos-1] > d2){ kd[pos]=kd[pos-1]; ki[pos]=ki[pos-1]; pos--; }
        kd[pos] = d2; ki[pos] = j;
      }
    }
  }
  size_t base = (size_t)gid*K;
  for (int k = 0; k < K; k++){
    nidx[base+k]  = (k < cnt) ? ki[k] : 0;
    valid[base+k] = (k < cnt) ? 1 : 0;
  }
}

// ---------------- edge-feature matrix build: [xj, pj-qi] f16, K-padded ----------------
__global__ void k_build_edge(const float* __restrict__ x, int Cx, int Nsrc,
                             const float* __restrict__ p, const float* __restrict__ q,
                             const int* __restrict__ nidx, int M, int K, int Cpad,
                             half_t* __restrict__ H){
  int i = blockIdx.x*blockDim.x + threadIdx.x;
  int total = 16*M*K*Cpad;
  if (i >= total) return;
  int c = i % Cpad;
  int r = i / Cpad;
  int b = r / (M*K);
  int rem = r - b*(M*K);
  int m = rem / K;
  int j = nidx[r];
  float v = 0.f;
  if (c < Cx)          v = x[((size_t)b*Nsrc + j)*Cx + c];
  else if (c < Cx + 3) v = p[((size_t)b*Nsrc + j)*3 + (c - Cx)]
                         - q[((size_t)b*M + m)*3 + (c - Cx)];
  H[i] = (half_t)v;
}

__global__ void k_build_sa4(const float* __restrict__ x3, const float* __restrict__ q3,
                            half_t* __restrict__ H){
  const int Cp = 1056;
  int i = blockIdx.x*blockDim.x + threadIdx.x;
  if (i >= 656*Cp) return;
  int r = i / Cp, c = i - r*Cp;
  float v = 0.f;
  if (c < 1024)      v = x3[(size_t)r*1024 + c];
  else if (c < 1027) v = q3[(size_t)r*3 + (c - 1024)];
  H[i] = (half_t)v;
}

// ---------------- WMMA GEMM core (wave-level 16x16 tile, K in chunks of 32) ----------------
DEV_INLINE v16h load_frag(const half_t* base){
  v8h lo = *(const v8h*)(base);       // K 0..7  (this half)
  v8h hi = *(const v8h*)(base + 16);  // K 16..23 (this half)
  return __builtin_shufflevector(lo, hi, 0,1,2,3,4,5,6,7,8,9,10,11,12,13,14,15);
}

DEV_INLINE v8f wave_tile_gemm(const half_t* __restrict__ A, const half_t* __restrict__ Wt,
                              int Kpad, int rowBase, int colBase, int lane){
  int hi8  = (lane >> 4) & 1;
  int rsel = lane & 15;
  const half_t* ap = A  + (size_t)(rowBase + rsel)*Kpad + hi8*8;
  const half_t* bp = Wt + (size_t)(colBase + rsel)*Kpad + hi8*8;
  v8f acc = {};
  for (int kc = 0; kc < Kpad; kc += 32){
    if (kc + 32 < Kpad) __builtin_prefetch(ap + kc + 32, 0, 1); // global_prefetch_b8
    v16h a = load_frag(ap + kc);
    v16h b = load_frag(bp + kc);
    acc = __builtin_amdgcn_wmma_f32_16x16x32_f16(false, a, false, b, (short)0, acc,
                                                 false, false);
  }
  return acc;
}

// pass 1: masked per-channel sum / sumsq for training-mode BN (no Y store)
__global__ void k_gemm_stats(const half_t* __restrict__ A, const half_t* __restrict__ Wt,
                             const float* __restrict__ bias,
                             const unsigned char* __restrict__ valid,
                             float* __restrict__ stats, int rows, int N, int Kpad){
  int lane = threadIdx.x & 31, wave = threadIdx.x >> 5;
  int tiles_n = N >> 4;
  int tiles   = (rows >> 4) * tiles_n;
  int tile = blockIdx.x*8 + wave;
  if (tile >= tiles) return;                 // whole-wave exit: EXEC stays all-ones
  int tm = tile / tiles_n, tn = tile - tm*tiles_n;
  v8f acc = wave_tile_gemm(A, Wt, Kpad, tm*16, tn*16, lane);
  int col   = tn*16 + (lane & 15);
  int rbase = tm*16 + ((lane >> 4) & 1)*8;
  float bv = bias[col];
  float s = 0.f, s2 = 0.f;
#pragma unroll
  for (int v = 0; v < 8; v++){
    float y = acc[v] + bv;
    float w = valid ? (valid[rbase + v] ? 1.f : 0.f) : 1.f;
    s  += w*y;
    s2 += w*y*y;
  }
  atomicAdd(&stats[col], s);
  atomicAdd(&stats[2048 + col], s2);
}

// pass 2: recompute tile, apply BN(gamma,beta)+ReLU, emit f16 A-matrix for next layer
__global__ void k_gemm_bn(const half_t* __restrict__ A, const half_t* __restrict__ Wt,
                          const float* __restrict__ bias,
                          const float* __restrict__ gamma, const float* __restrict__ beta,
                          const float* __restrict__ stats, int rows, int N, int Kpad,
                          half_t* __restrict__ out, int ldo){
  int lane = threadIdx.x & 31, wave = threadIdx.x >> 5;
  int tiles_n = N >> 4;
  int tiles   = (rows >> 4) * tiles_n;
  int tile = blockIdx.x*8 + wave;
  if (tile >= tiles) return;
  int tm = tile / tiles_n, tn = tile - tm*tiles_n;
  v8f acc = wave_tile_gemm(A, Wt, Kpad, tm*16, tn*16, lane);
  int col   = tn*16 + (lane & 15);
  int rbase = tm*16 + ((lane >> 4) & 1)*8;
  float n    = fmaxf(stats[4096], 1.f);
  float mean = stats[col] / n;
  float var  = fmaxf(stats[2048 + col] / n - mean*mean, 0.f);
  float sc   = gamma[col] * rsqrtf(var + 1e-5f);
  float sh   = beta[col] - sc*mean;
  float bv   = bias[col];
#pragma unroll
  for (int v = 0; v < 8; v++){
    float y = fmaxf(sc*(acc[v] + bv) + sh, 0.f);
    out[(size_t)(rbase + v)*ldo + col] = (half_t)y;
  }
}

// ---------------- neighbor max-aggregation (masked; empty -> 0) ----------------
__global__ void k_aggmax(const half_t* __restrict__ H, const unsigned char* __restrict__ valid,
                         int M, int K, int C, float* __restrict__ xout, int Ctot, int coff){
  int i = blockIdx.x*blockDim.x + threadIdx.x;
  int total = 16*M*C;
  if (i >= total) return;
  int c = i % C;
  int bm = i / C;
  size_t ebase = (size_t)bm*K;
  float best = -3.4e38f; int any = 0;
  for (int k = 0; k < K; k++){
    if (valid[ebase + k]){
      best = fmaxf(best, (float)H[(ebase + k)*(size_t)C + c]);
      any = 1;
    }
  }
  xout[(size_t)bm*Ctot + coff + c] = any ? best : 0.f;
}

// global max pool over 41 nodes -> f16 [16,2048] head input
__global__ void k_gpool(const half_t* __restrict__ Hin, half_t* __restrict__ Hout){
  int i = blockIdx.x*blockDim.x + threadIdx.x;
  if (i >= 16*2048) return;
  int b = i >> 11, c = i & 2047;
  float best = -3.4e38f;
  for (int m = 0; m < 41; m++)
    best = fmaxf(best, (float)Hin[((size_t)(b*41 + m))*2048 + c]);
  Hout[i] = (half_t)best;
}

// final linear head: [16,64] @ [64,5] + b  (no norm/act), f32 out
__global__ void k_head3(const half_t* __restrict__ H, const float* __restrict__ W,
                        const float* __restrict__ bias, float* __restrict__ out){
  int i = threadIdx.x;
  if (i >= 16*5) return;
  int b = i / 5, o = i - b*5;
  float s = bias[o];
  for (int c = 0; c < 64; c++) s += (float)H[b*64 + c] * W[c*5 + o];
  out[i] = s;
}

// ---------------- host orchestration ----------------
extern "C" void kernel_launch(void* const* d_in, const int* in_sizes, int n_in,
                              void* d_out, int out_size, void* d_ws, size_t ws_size,
                              hipStream_t stream) {
  (void)in_sizes; (void)n_in; (void)out_size;
  if (ws_size < WS_NEED) return;

  const float* pos = (const float*)d_in[0];
  char* ws = (char*)d_ws;
  half_t* WT   = (half_t*)(ws + OFF_WT);
  int*    idx1 = (int*)  (ws + OFF_IDX1);
  float*  q1   = (float*)(ws + OFF_Q1);
  int*    idx2 = (int*)  (ws + OFF_IDX2);
  float*  q2   = (float*)(ws + OFF_Q2);
  int*    idx3 = (int*)  (ws + OFF_IDX3);
  float*  q3   = (float*)(ws + OFF_Q3);
  float*  x1   = (float*)(ws + OFF_X1);
  float*  x2   = (float*)(ws + OFF_X2);
  float*  x3   = (float*)(ws + OFF_X3);
  int*    nidx = (int*)  (ws + OFF_NIDX);
  unsigned char* valid = (unsigned char*)(ws + OFF_VALID);
  float*  stats = (float*)(ws + OFF_STATS);
  half_t* Ha = (half_t*)(ws + OFF_HA);
  half_t* Hb = (half_t*)(ws + OFF_HB);

  // ---- convert & transpose all weights to f16 [Nout, Kpad] ----
  struct WtCfg { int widx; size_t off; int Kin, Nout, Kpad; };
  const WtCfg wc[10] = {
    {1,  O_W10, 6,    16,   32},
    {5,  O_W11, 16,   32,   32},
    {9,  O_W20, 67,   64,   96},
    {13, O_W21, 64,   128,  64},
    {17, O_W30, 259,  256,  288},
    {21, O_W31, 256,  512,  256},
    {25, O_W40, 1027, 1024, 1056},
    {29, O_W41, 1024, 2048, 1024},
    {33, O_H1,  2048, 512,  2048},
    {37, O_H2,  512,  64,   512},
  };
  for (int i = 0; i < 10; i++){
    int tot = wc[i].Nout * wc[i].Kpad;
    k_wt_conv<<<CDIV(tot,256),256,0,stream>>>((const float*)d_in[wc[i].widx],
                                              WT + wc[i].off, wc[i].Kin, wc[i].Nout, wc[i].Kpad);
  }

  // GEMM layer = zero stats, stats pass, BN pass; widx indexes (W,b,g,be)
  auto gemm_layer = [&](const half_t* A, const half_t* Wt, int widx,
                        const unsigned char* vmask, int rows, int N, int Kpad,
                        half_t* outp, int ldo){
    const float* bb = (const float*)d_in[widx+1];
    const float* gg = (const float*)d_in[widx+2];
    const float* be = (const float*)d_in[widx+3];
    int blocks = CDIV((rows/16)*(N/16), 8);
    k_zero_f32<<<CDIV(4096,256),256,0,stream>>>(stats, 4096);
    k_gemm_stats<<<blocks,256,0,stream>>>(A, Wt, bb, vmask, stats, rows, N, Kpad);
    k_gemm_bn<<<blocks,256,0,stream>>>(A, Wt, bb, gg, be, stats, rows, N, Kpad, outp, ldo);
  };

  // one radius branch of an SA stage: ballq -> edges -> MLP(2 layers, masked BN) -> max-agg
  auto run_branch = [&](const float* xin, int Cx, const float* psrc, int Nsrc,
                        const float* qq, int M, float r, int K, int Cpad,
                        size_t w0off, int w0idx, int Nout0, int ldo0,
                        size_t w1off, int w1idx, int Nout1,
                        float* xout, int Ctot, int coff){
    int rows = 16*M*K;
    k_ballq<<<CDIV(16*M,256),256,0,stream>>>(qq, psrc, M, Nsrc, r*r, K, nidx, valid);
    k_zero_f32<<<CDIV(4098,256),256,0,stream>>>(stats, 4098);
    k_count_valid<<<CDIV(rows,256),256,0,stream>>>(valid, rows, stats + 4096);
    k_build_edge<<<CDIV(rows*Cpad,256),256,0,stream>>>(xin, Cx, Nsrc, psrc, qq,
                                                       nidx, M, K, Cpad, Ha);
    if (ldo0 > Nout0)  // pad cols of next A-matrix (only sa1 L0: 16 -> 32)
      k_zero_f16<<<CDIV(rows*ldo0,256),256,0,stream>>>(Hb, rows*ldo0);
    gemm_layer(Ha, WT + w0off, w0idx, valid, rows, Nout0, Cpad, Hb, ldo0);
    gemm_layer(Hb, WT + w1off, w1idx, valid, rows, Nout1, ldo0, Ha, Nout1);
    k_aggmax<<<CDIV(16*M*Nout1,256),256,0,stream>>>(Ha, valid, M, K, Nout1, xout, Ctot, coff);
  };

  // ---- SA stage 1: 4096 -> 410, r={0.05,0.1}, K=16, MLP 6->16->32 ----
  k_fps<<<16,256,0,stream>>>(pos, 4096, 410, idx1, q1);
  run_branch(pos, 3, pos, 4096, q1, 410, 0.05f, 16, 32, O_W10, 1, 16, 32, O_W11, 5, 32, x1, 64, 0);
  run_branch(pos, 3, pos, 4096, q1, 410, 0.10f, 16, 32, O_W10, 1, 16, 32, O_W11, 5, 32, x1, 64, 32);

  // ---- SA stage 2: 410 -> 82, r={0.2,0.4}, K=32, MLP 67->64->128 ----
  k_fps<<<16,256,0,stream>>>(q1, 410, 82, idx2, q2);
  run_branch(x1, 64, q1, 410, q2, 82, 0.2f, 32, 96, O_W20, 9, 64, 64, O_W21, 13, 128, x2, 256, 0);
  run_branch(x1, 64, q1, 410, q2, 82, 0.4f, 32, 96, O_W20, 9, 64, 64, O_W21, 13, 128, x2, 256, 128);

  // ---- SA stage 3: 82 -> 41, r={0.5,1.0}, K=64, MLP 259->256->512 ----
  k_fps<<<16,256,0,stream>>>(q2, 82, 41, idx3, q3);
  run_branch(x2, 256, q2, 82, q3, 41, 0.5f, 64, 288, O_W30, 17, 256, 256, O_W31, 21, 512, x3, 1024, 0);
  run_branch(x2, 256, q2, 82, q3, 41, 1.0f, 64, 288, O_W30, 17, 256, 256, O_W31, 21, 512, x3, 1024, 512);

  // ---- SA stage 4 (global MLP 1027->1024->2048 over 656 rows, unmasked BN) ----
  k_build_sa4<<<CDIV(656*1056,256),256,0,stream>>>(x3, q3, Ha);
  k_set_f32<<<1,32,0,stream>>>(stats + 4096, 656.f);
  gemm_layer(Ha, WT + O_W40, 25, nullptr, 656, 1024, 1056, Hb, 1024);
  gemm_layer(Hb, WT + O_W41, 29, nullptr, 656, 2048, 1024, Ha, 2048);

  // ---- global max pool + heads ----
  k_gpool<<<CDIV(16*2048,256),256,0,stream>>>(Ha, Hb);
  k_set_f32<<<1,32,0,stream>>>(stats + 4096, 16.f);
  gemm_layer(Hb, WT + O_H1, 33, nullptr, 16, 512, 2048, Ha, 512);
  gemm_layer(Ha, WT + O_H2, 37, nullptr, 16, 64, 512, Hb, 64);
  k_head3<<<1,128,0,stream>>>(Hb, (const float*)d_in[41], (const float*)d_in[42], (float*)d_out);
}
